// Block_773094113529
// MI455X (gfx1250) — matrix operand order repaired and checked by reading, not compiled, not run on hardware
//
#include <hip/hip_runtime.h>
#include <hip/hip_bf16.h>

// ---------------------------------------------------------------------------
// Transformer block (B=4,S=1024,D=1024,H=16,HD=64) on MI455X / gfx1250.
// Compute-bound (~120 GFLOP) -> all GEMMs on v_wmma_f32_16x16x32_bf16.
// CDNA5 data movement:
//   * GEMM B tiles staged via TDM (tensor_load_to_lds + s_wait_tensorcnt)
//   * attention K^T/V tiles via GLOBAL_LOAD_ASYNC_TO_LDS_B128 (s_wait_asynccnt)
// K projection is written pre-transposed [B,H,HD,S] so every K^T LDS fill is a
// contiguous 16B async copy (no u16 transpose scatter).
// Attention BN1 via two-pass flash scheme.  Workspace ~140 MB.
// ---------------------------------------------------------------------------

#define BB 4
#define SS 1024
#define DD 1024
#define HH 16
#define HD 64

typedef __attribute__((ext_vector_type(16))) __bf16 v16bf;
typedef __attribute__((ext_vector_type(8)))  __bf16 v8bf;
typedef __attribute__((ext_vector_type(8)))  float  v8f;
typedef __attribute__((ext_vector_type(4)))  unsigned u32x4;
typedef __attribute__((ext_vector_type(8)))  unsigned u32x8;

__device__ __forceinline__ __bf16 f2bf(float f) {
  unsigned u = __float_as_uint(f);
  unsigned r = (u + 0x7FFFu + ((u >> 16) & 1u)) >> 16;
  return __builtin_bit_cast(__bf16, (unsigned short)r);
}

__device__ __forceinline__ v8f wmma_bf16(v16bf a, v16bf b, v8f c) {
  return __builtin_amdgcn_wmma_f32_16x16x32_bf16(
      false, a, false, b, (short)0, c, false, false);
}

// ---- CDNA5 async global->LDS copy (ASYNCcnt-tracked) ----------------------
__device__ __forceinline__ void async_g2l(unsigned lds_off, const void* gptr) {
  asm volatile("global_load_async_to_lds_b128 %0, %1, off"
               :: "v"(lds_off), "v"(gptr)
               : "memory");
}
template <int N>
__device__ __forceinline__ void wait_async() {
  asm volatile("s_wait_asynccnt %0" :: "i"(N) : "memory");
}
__device__ __forceinline__ unsigned lds_off32(const void* p) {
  return (unsigned)(uintptr_t)p;   // AS3->generic keeps LDS offset in low 32b
}

// ---- CDNA5 Tensor Data Mover: 2-D tile global->LDS (TENSORcnt-tracked) ----
// D# group0 (ISA 8.3): [1:0]=count=1, [63:32]=lds_addr, [120:64]=global_addr,
// [127:126]=type=2.  group1 (ISA 8.4): data_size=2B, tensor_dim0/1,
// tile_dim0/1, tensor_dim0_stride.
__device__ __forceinline__ u32x4 tdm_g0(unsigned lds_addr, const void* gaddr) {
  unsigned long long a = (unsigned long long)(uintptr_t)gaddr;
  u32x4 g;
  g[0] = 1u;                                   // count=1, user descriptor
  g[1] = lds_addr;
  g[2] = (unsigned)a;                          // global_addr[31:0]
  g[3] = (unsigned)(a >> 32) | (2u << 30);     // global_addr[56:32] | type=2
  return g;
}
__device__ __forceinline__ u32x8 tdm_g1(unsigned dim0, unsigned dim1,
                                        unsigned tile0, unsigned tile1,
                                        unsigned long long stride0) {
  u32x8 g;
  g[0] = 1u << 16;                             // data_size = 2 bytes
  g[1] = (dim0 & 0xFFFFu) << 16;               // tensor_dim0[15:0]
  g[2] = (dim0 >> 16) | ((dim1 & 0xFFFFu) << 16);
  g[3] = (dim1 >> 16) | (tile0 << 16);         // tile_dim0
  g[4] = tile1;                                // tile_dim1 (tile_dim2=0)
  g[5] = (unsigned)stride0;                    // tensor_dim0_stride[31:0]
  g[6] = (unsigned)(stride0 >> 32) & 0xFFFFu;  // stride0[47:32]
  g[7] = 0u;
  return g;
}
__device__ __forceinline__ void tdm_load(u32x4 g0, u32x8 g1) {
  asm volatile("tensor_load_to_lds %0, %1" :: "s"(g0), "s"(g1) : "memory");
}

// A fragment (16x32 bf16, row-major, stride lda).
__device__ __forceinline__ v16bf load_a_frag(const __bf16* A, int lda,
                                             int row0, int k0, int lane) {
  int r  = lane & 15;
  int kh = (lane >> 4) << 3;
  const __bf16* p = A + (size_t)(row0 + r) * lda + k0 + kh;
  v8bf lo = *(const v8bf*)p;
  v8bf hi = *(const v8bf*)(p + 16);
  v16bf out;
#pragma unroll
  for (int i = 0; i < 8; ++i) { out[i] = lo[i]; out[8 + i] = hi[i]; }
  return out;
}

// B fragment (32x16) from row-major [32 x 64] tile: lane = N column, K in table order.
__device__ __forceinline__ v16bf load_b_frag64(const __bf16* Bm, int n0, int lane) {
  int n = n0 + (lane & 15);
  int kh = (lane >> 4) << 3;
  v16bf out;
#pragma unroll
  for (int i = 0; i < 8; ++i) {
    out[i] = Bm[(kh + i) * 64 + n];
    out[8 + i] = Bm[(16 + kh + i) * 64 + n];
  }
  return out;
}

// B fragment from K^T LDS tile [64(d) x 32(kcol)]: lane = kcol, d in table order.
__device__ __forceinline__ v16bf ldsKT_frag(const __bf16* kt, int d0, int kloc16, int lane) {
  int n = kloc16 + (lane & 15);
  int dh = d0 + ((lane >> 4) << 3);
  v16bf out;
#pragma unroll
  for (int i = 0; i < 8; ++i) {
    out[i] = kt[(dh + i) * 32 + n];
    out[8 + i] = kt[(dh + 16 + i) * 32 + n];
  }
  return out;
}

// --------------------------- small utility kernels -------------------------

__global__ void zero_f32(float* p, int n) {
  int i = blockIdx.x * 256 + threadIdx.x;
  if (i < n) p[i] = 0.0f;
}

__global__ void cast_bf16_kernel(const float* __restrict__ s,
                                 __bf16* __restrict__ d, int n) {
  int i = blockIdx.x * 256 + threadIdx.x;
  if (i < n) d[i] = f2bf(s[i]);
}

__global__ void ln_cast(const float* __restrict__ X, const float* __restrict__ g,
                        const float* __restrict__ b, __bf16* __restrict__ Y) {
  int row = blockIdx.x;
  const float* x = X + (size_t)row * DD;
  float v[4], s = 0.f, s2 = 0.f;
#pragma unroll
  for (int i = 0; i < 4; ++i) {
    v[i] = x[threadIdx.x + 256 * i];
    s += v[i]; s2 += v[i] * v[i];
  }
#pragma unroll
  for (int o = 16; o > 0; o >>= 1) {
    s  += __shfl_down(s,  o, 32);
    s2 += __shfl_down(s2, o, 32);
  }
  __shared__ float red[20];
  int lane = threadIdx.x & 31, w = threadIdx.x >> 5;
  if (lane == 0) { red[w] = s; red[8 + w] = s2; }
  __syncthreads();
  if (threadIdx.x == 0) {
    float ts = 0.f, t2 = 0.f;
    for (int i = 0; i < 8; ++i) { ts += red[i]; t2 += red[8 + i]; }
    float m = ts * (1.0f / DD);
    red[16] = m;
    red[17] = rsqrtf(t2 * (1.0f / DD) - m * m + 1e-6f);
  }
  __syncthreads();
  float m = red[16], inv = red[17];
#pragma unroll
  for (int i = 0; i < 4; ++i) {
    int c = threadIdx.x + 256 * i;
    Y[(size_t)row * DD + c] = f2bf((v[i] - m) * inv * g[c] + b[c]);
  }
}

// ------------------------------- main GEMM ---------------------------------
// Block = 4 waves = 64(M) x 64(N).  B tile (32x64) staged by TDM, double
// buffered.  MODE: 0=QKV->bf16 [B,H,S,HD], 1=+resid->f32, 2=GELU->bf16,
// 3=+resid->f32, 4=K^T -> bf16 [B,H,HD,S].
template <int MODE>
__global__ void gemm_wmma(const __bf16* __restrict__ A, const __bf16* __restrict__ W,
                          const float* __restrict__ bias,
                          const float* __restrict__ resid,
                          float* __restrict__ outf, __bf16* __restrict__ outb,
                          int M, int N, int K) {
  __shared__ __align__(32) __bf16 bt[2][32][64];     // 8 KB double buffer
  const int tid = threadIdx.x;
  const int lane = tid & 31, wave = tid >> 5;
  const int m0 = blockIdx.x * 64 + wave * 16;
  const int n0 = blockIdx.y * 64;

  const unsigned ldsBase = lds_off32(&bt[0][0][0]);
  const unsigned bufB = 32 * 64 * 2;
  const u32x8 g1 = tdm_g1(/*dim0=*/(unsigned)N, /*dim1=*/(unsigned)K,
                          /*tile0=*/64u, /*tile1=*/32u, /*stride0=*/(unsigned long long)N);
  const int T = K / 32;

  if (wave == 0) {   // one TDM descriptor moves the whole B tile
    tdm_load(tdm_g0(ldsBase,        W + (size_t)0 * N + n0), g1);
    tdm_load(tdm_g0(ldsBase + bufB, W + (size_t)32 * N + n0), g1);
  }

  v8f c[4] = {v8f{}, v8f{}, v8f{}, v8f{}};
  for (int t = 0; t < T; ++t) {
    if (wave == 0) {
      if (t + 1 < T) __builtin_amdgcn_s_wait_tensorcnt(1);
      else           __builtin_amdgcn_s_wait_tensorcnt(0);
    }
    __syncthreads();
    const int pb = t & 1;
    if (t + 4 < T)
      __builtin_prefetch(A + (size_t)(m0 + (lane & 15)) * K + (t + 4) * 32, 0, 0);
    v16bf a = load_a_frag(A, K, m0, t * 32, lane);
#pragma unroll
    for (int j = 0; j < 4; ++j) {
      v16bf b = load_b_frag64(&bt[pb][0][0], 16 * j, lane);
      c[j] = wmma_bf16(a, b, c[j]);
    }
    __syncthreads();
    if (wave == 0 && t + 2 < T)
      tdm_load(tdm_g0(ldsBase + (unsigned)pb * bufB,
                      W + (size_t)(t + 2) * 32 * N + n0), g1);
  }

  const int col  = lane & 15;
  const int roff = (lane >> 4) << 3;
#pragma unroll
  for (int j = 0; j < 4; ++j) {
#pragma unroll
    for (int i = 0; i < 8; ++i) {
      int m = m0 + i + roff;
      int n = n0 + 16 * j + col;
      float v = c[j][i] + bias[n];
      if (MODE == 0) {
        int b_ = m >> 10, s_ = m & 1023, h_ = n >> 6, hd = n & 63;
        outb[((((size_t)b_ * HH + h_) << 10) + s_) * HD + hd] = f2bf(v);
      } else if (MODE == 4) {      // K^T: [B,H,HD,S]
        int b_ = m >> 10, s_ = m & 1023, h_ = n >> 6, hd = n & 63;
        outb[((((size_t)b_ * HH + h_) * HD + hd) << 10) + s_] = f2bf(v);
      } else if (MODE == 1) {
        outf[(size_t)m * N + n] = v + resid[(size_t)m * N + n];
      } else if (MODE == 2) {
        float gl = 0.5f * v * (1.0f + erff(v * 0.70710678118f));
        outb[(size_t)m * N + n] = f2bf(gl);
      } else {
        outf[(size_t)m * N + n] = v + resid[(size_t)m * N + n];
      }
    }
  }
}

// ------------------- attention pass 1: score statistics --------------------
// 4 waves share (b,h); K^T chunks [64(d) x 32(kcol)] async-staged.
__global__ void score_stats(const __bf16* __restrict__ Q, const __bf16* __restrict__ KT,
                            float* __restrict__ sums, float* __restrict__ sqs) {
  __shared__ __align__(32) __bf16 kt2[2][64][32];
  const int tid = threadIdx.x;
  const int lane = tid & 31, wave = tid >> 5;
  const int qt = blockIdx.x * 4 + wave;
  const int h = blockIdx.y, bz = blockIdx.z;
  const size_t base = (((size_t)bz * HH + h) << 10) * HD;   // == (..)*HD*SS
  const __bf16* q  = Q + base;
  const __bf16* kT = KT + base;

  // copy layout: thread -> (d row, 16-elem chunk); 2 async b128 per tile
  const int kd  = tid >> 1;            // 0..63
  const int kco = (tid & 1) << 4;      // 0 / 16
  const __bf16* ksrc = kT + (size_t)kd * SS + kco;
  const unsigned lk0 = lds_off32(&kt2[0][kd][kco]);
  const unsigned bufB = 64 * 32 * 2;

  v16bf aq0 = load_a_frag(q, HD, qt * 16, 0,  lane);
  v16bf aq1 = load_a_frag(q, HD, qt * 16, 32, lane);

  const int T = SS / 32;
  async_g2l(lk0, ksrc);          async_g2l(lk0 + 16, ksrc + 8);
  async_g2l(lk0 + bufB, ksrc + 32);  async_g2l(lk0 + bufB + 16, ksrc + 40);

  float ls = 0.f, ls2 = 0.f;
  for (int t = 0; t < T; ++t) {
    if (t + 1 < T) wait_async<2>(); else wait_async<0>();
    __syncthreads();
    const int pb = t & 1;
    const __bf16* ktile = &kt2[pb][0][0];
#pragma unroll
    for (int kt = 0; kt < 2; ++kt) {
      v8f c = wmma_bf16(aq0, ldsKT_frag(ktile, 0,  kt * 16, lane), v8f{});
      c     = wmma_bf16(aq1, ldsKT_frag(ktile, 32, kt * 16, lane), c);
#pragma unroll
      for (int i = 0; i < 8; ++i) {
        float s = c[i] * 0.125f;
        ls += s; ls2 += s * s;
      }
    }
    __syncthreads();
    if (t + 2 < T) {
      const __bf16* ks2 = ksrc + (size_t)(t + 2) * 32;
      const unsigned lo = (unsigned)pb * bufB;
      async_g2l(lk0 + lo, ks2);
      async_g2l(lk0 + lo + 16, ks2 + 8);
    }
  }
#pragma unroll
  for (int o = 16; o > 0; o >>= 1) {
    ls  += __shfl_down(ls,  o, 32);
    ls2 += __shfl_down(ls2, o, 32);
  }
  __shared__ float red[8];
  if (lane == 0) { red[wave] = ls; red[4 + wave] = ls2; }
  __syncthreads();
  if (threadIdx.x == 0) {
    atomicAdd(&sums[h], red[0] + red[1] + red[2] + red[3]);
    atomicAdd(&sqs[h],  red[4] + red[5] + red[6] + red[7]);
  }
}

__global__ void bn_finalize(const float* __restrict__ sums, const float* __restrict__ sqs,
                            const float* __restrict__ g, const float* __restrict__ bt,
                            float count_inv, float* __restrict__ ss) {
  int h = threadIdx.x;
  if (h < HH) {
    float m   = sums[h] * count_inv;
    float var = sqs[h] * count_inv - m * m;
    float inv = rsqrtf(var + 1e-5f);
    float sc  = inv * g[h];
    ss[h]      = sc;
    ss[HH + h] = bt[h] - m * sc;
  }
}

// ------------- attention pass 2: BN + online softmax + probs@V -------------
__global__ void attn_softmax_ctx(const __bf16* __restrict__ Q, const __bf16* __restrict__ KT,
                                 const __bf16* __restrict__ V,
                                 const float* __restrict__ bn1_ss,
                                 float* __restrict__ ctx) {
  __shared__ __align__(32) __bf16 kt2[2][64][32];
  __shared__ __align__(32) __bf16 vt2[2][32][64];
  __shared__ __align__(32) __bf16 plds[4][16][32];
  const int tid = threadIdx.x;
  const int lane = tid & 31, wave = tid >> 5;
  const int qt = blockIdx.x * 4 + wave;
  const int h = blockIdx.y, bz = blockIdx.z;
  const size_t base = (((size_t)bz * HH + h) << 10) * HD;
  const __bf16* q  = Q + base;
  const __bf16* kT = KT + base;
  const __bf16* v  = V + base;

  const float sc = bn1_ss[h] * 0.125f;
  const float sh = bn1_ss[HH + h];
  const int half = lane >> 4;
  const int col  = lane & 15;

  const int kd  = tid >> 1;
  const int kco = (tid & 1) << 4;
  const __bf16* ksrc = kT + (size_t)kd * SS + kco;
  const unsigned lk0 = lds_off32(&kt2[0][kd][kco]);
  const unsigned bufK = 64 * 32 * 2;

  const int crow = tid >> 3;
  const int ccol = (tid & 7) * 8;
  const __bf16* vsrc = v + (size_t)crow * HD + ccol;
  const unsigned lv0 = lds_off32(&vt2[0][crow][ccol]);
  const unsigned lv1 = lds_off32(&vt2[0][crow + 16][ccol]);
  const unsigned bufV = 32 * 64 * 2;
  const size_t rstep = (size_t)16 * HD, kstep = (size_t)32 * HD;

  v16bf aq0 = load_a_frag(q, HD, qt * 16, 0,  lane);
  v16bf aq1 = load_a_frag(q, HD, qt * 16, 32, lane);

  float mrun[8], lrun[8];
#pragma unroll
  for (int i = 0; i < 8; ++i) { mrun[i] = -3.0e38f; lrun[i] = 0.f; }
  v8f acc[4] = {v8f{}, v8f{}, v8f{}, v8f{}};

  const int T = SS / 32;
  async_g2l(lk0, ksrc);              async_g2l(lk0 + 16, ksrc + 8);
  async_g2l(lv0, vsrc);              async_g2l(lv1, vsrc + rstep);
  async_g2l(lk0 + bufK, ksrc + 32);  async_g2l(lk0 + bufK + 16, ksrc + 40);
  async_g2l(lv0 + bufV, vsrc + kstep);
  async_g2l(lv1 + bufV, vsrc + kstep + rstep);

  for (int t = 0; t < T; ++t) {
    if (t + 1 < T) wait_async<4>(); else wait_async<0>();
    __syncthreads();
    const int pb = t & 1;
    const __bf16* ktile = &kt2[pb][0][0];
#pragma unroll
    for (int kt = 0; kt < 2; ++kt) {
      v8f c = wmma_bf16(aq0, ldsKT_frag(ktile, 0,  kt * 16, lane), v8f{});
      c     = wmma_bf16(aq1, ldsKT_frag(ktile, 32, kt * 16, lane), c);
      float rescale[8];
#pragma unroll
      for (int i = 0; i < 8; ++i) {
        float s = c[i] * sc + sh;
        float mx = s;
#pragma unroll
        for (int o = 1; o < 16; o <<= 1) mx = fmaxf(mx, __shfl_xor(mx, o, 32));
        float newm = fmaxf(mrun[i], mx);
        rescale[i] = __expf(mrun[i] - newm);
        float p = __expf(s - newm);
        float ps = p;
#pragma unroll
        for (int o = 1; o < 16; o <<= 1) ps += __shfl_xor(ps, o, 32);
        lrun[i] = lrun[i] * rescale[i] + ps;
        mrun[i] = newm;
        plds[wave][i + 8 * half][col + kt * 16] = f2bf(p);
      }
#pragma unroll
      for (int j = 0; j < 4; ++j)
#pragma unroll
        for (int i = 0; i < 8; ++i) acc[j][i] *= rescale[i];
    }
    v16bf pa = load_a_frag(&plds[wave][0][0], 32, 0, 0, lane);
#pragma unroll
    for (int j = 0; j < 4; ++j) {
      v16bf vbf = load_b_frag64(&vt2[pb][0][0], 16 * j, lane);
      acc[j] = wmma_bf16(pa, vbf, acc[j]);
    }
    __syncthreads();
    if (t + 2 < T) {
      const __bf16* ks2 = ksrc + (size_t)(t + 2) * 32;
      const __bf16* vs2 = vsrc + (size_t)(t + 2) * kstep;
      const unsigned lok = (unsigned)pb * bufK;
      const unsigned lov = (unsigned)pb * bufV;
      async_g2l(lk0 + lok, ks2);  async_g2l(lk0 + lok + 16, ks2 + 8);
      async_g2l(lv0 + lov, vs2);  async_g2l(lv1 + lov, vs2 + rstep);
    }
  }
#pragma unroll
  for (int j = 0; j < 4; ++j)
#pragma unroll
    for (int i = 0; i < 8; ++i) {
      int row = qt * 16 + i + 8 * half;
      ctx[base + (size_t)row * HD + j * 16 + col] = acc[j][i] / lrun[i];
    }
}

// BN2 statistics over materialized ctx.
__global__ void ctx_stats(const float* __restrict__ ctx, float* __restrict__ sums,
                          float* __restrict__ sqs) {
  size_t base = (size_t)blockIdx.x * 1024;
  int h = (int)((base >> 16) & (HH - 1));
  float s = 0.f, s2 = 0.f;
#pragma unroll
  for (int i = 0; i < 4; ++i) {
    float v = ctx[base + threadIdx.x + 256 * i];
    s += v; s2 += v * v;
  }
#pragma unroll
  for (int o = 16; o > 0; o >>= 1) {
    s  += __shfl_down(s,  o, 32);
    s2 += __shfl_down(s2, o, 32);
  }
  __shared__ float red[16];
  int lane = threadIdx.x & 31, w = threadIdx.x >> 5;
  if (lane == 0) { red[w] = s; red[8 + w] = s2; }
  __syncthreads();
  if (threadIdx.x == 0) {
    float a = 0.f, a2 = 0.f;
    for (int i = 0; i < 8; ++i) { a += red[i]; a2 += red[8 + i]; }
    atomicAdd(&sums[h], a); atomicAdd(&sqs[h], a2);
  }
}

__global__ void ctx_norm_transpose(const float* __restrict__ ctx,
                                   const float* __restrict__ ss,
                                   __bf16* __restrict__ out) {
  size_t idx = (size_t)blockIdx.x * 256 + threadIdx.x;
  int hd = idx & 63;
  size_t t = idx >> 6;
  int s_ = t & 1023; t >>= 10;
  int h = t & (HH - 1); int b = (int)(t >> 4);
  float v = ctx[idx] * ss[h] + ss[HH + h];
  out[((size_t)(b * SS + s_)) * DD + h * HD + hd] = f2bf(v);
}

// ---------------------------------------------------------------------------

extern "C" void kernel_launch(void* const* d_in, const int* in_sizes, int n_in,
                              void* d_out, int out_size, void* d_ws, size_t ws_size,
                              hipStream_t stream) {
  const float* x     = (const float*)d_in[0];
  const float* Wq    = (const float*)d_in[1];
  const float* bq    = (const float*)d_in[2];
  const float* Wk    = (const float*)d_in[3];
  const float* bk    = (const float*)d_in[4];
  const float* Wv    = (const float*)d_in[5];
  const float* bv    = (const float*)d_in[6];
  const float* Wo    = (const float*)d_in[7];
  const float* bo    = (const float*)d_in[8];
  const float* bn1_g = (const float*)d_in[9];
  const float* bn1_b = (const float*)d_in[10];
  const float* bn2_g = (const float*)d_in[11];
  const float* bn2_b = (const float*)d_in[12];
  const float* ln1_g = (const float*)d_in[13];
  const float* ln1_b = (const float*)d_in[14];
  const float* ln2_g = (const float*)d_in[15];
  const float* ln2_b = (const float*)d_in[16];
  const float* W1    = (const float*)d_in[17];
  const float* b1    = (const float*)d_in[18];
  const float* W2    = (const float*)d_in[19];
  const float* b2    = (const float*)d_in[20];
  float* out = (float*)d_out;

  const int M = BB * SS;
  char* wp = (char*)d_ws;
  auto carve = [&](size_t bytes) {
    void* p = (void*)wp;
    wp += (bytes + 255) & ~(size_t)255;
    return p;
  };
  __bf16* yb    = (__bf16*)carve((size_t)M * DD * 2);
  __bf16* wqb   = (__bf16*)carve((size_t)DD * DD * 2);
  __bf16* wkb   = (__bf16*)carve((size_t)DD * DD * 2);
  __bf16* wvb   = (__bf16*)carve((size_t)DD * DD * 2);
  __bf16* wob   = (__bf16*)carve((size_t)DD * DD * 2);
  __bf16* w1b   = (__bf16*)carve((size_t)DD * 4 * DD * 2);
  __bf16* w2b   = (__bf16*)carve((size_t)4 * DD * DD * 2);
  __bf16* qb    = (__bf16*)carve((size_t)M * DD * 2);
  __bf16* kTb   = (__bf16*)carve((size_t)M * DD * 2);   // [B,H,HD,S]
  __bf16* vb    = (__bf16*)carve((size_t)M * DD * 2);
  float*  ctx   = (float*) carve((size_t)M * DD * 4);
  __bf16* attnb = (__bf16*)carve((size_t)M * DD * 2);
  float*  x2    = (float*) carve((size_t)M * DD * 4);
  __bf16* y2b   = (__bf16*)carve((size_t)M * DD * 2);
  __bf16* h1b   = (__bf16*)carve((size_t)M * 4 * DD * 2);
  float*  stats = (float*) carve(128 * 4);
  float* bn1_sum = stats;        float* bn1_sq = stats + 16;
  float* bn1_ss  = stats + 32;
  float* bn2_sum = stats + 64;   float* bn2_sq = stats + 80;
  float* bn2_ss  = stats + 96;

  ln_cast<<<M, 256, 0, stream>>>(x, ln1_g, ln1_b, yb);
  cast_bf16_kernel<<<(DD * DD) / 256, 256, 0, stream>>>(Wq, wqb, DD * DD);
  cast_bf16_kernel<<<(DD * DD) / 256, 256, 0, stream>>>(Wk, wkb, DD * DD);
  cast_bf16_kernel<<<(DD * DD) / 256, 256, 0, stream>>>(Wv, wvb, DD * DD);
  cast_bf16_kernel<<<(DD * DD) / 256, 256, 0, stream>>>(Wo, wob, DD * DD);
  cast_bf16_kernel<<<(4 * DD * DD) / 256, 256, 0, stream>>>(W1, w1b, 4 * DD * DD);
  cast_bf16_kernel<<<(4 * DD * DD) / 256, 256, 0, stream>>>(W2, w2b, 4 * DD * DD);

  gemm_wmma<0><<<dim3(M / 64, DD / 64), 128, 0, stream>>>(yb, wqb, bq, nullptr, nullptr, qb, M, DD, DD);
  gemm_wmma<4><<<dim3(M / 64, DD / 64), 128, 0, stream>>>(yb, wkb, bk, nullptr, nullptr, kTb, M, DD, DD);
  gemm_wmma<0><<<dim3(M / 64, DD / 64), 128, 0, stream>>>(yb, wvb, bv, nullptr, nullptr, vb, M, DD, DD);

  zero_f32<<<1, 256, 0, stream>>>(stats, 128);
  score_stats<<<dim3(SS / 64, HH, BB), 128, 0, stream>>>(qb, kTb, bn1_sum, bn1_sq);
  bn_finalize<<<1, 32, 0, stream>>>(bn1_sum, bn1_sq, bn1_g, bn1_b,
                                    1.0f / ((float)BB * SS * SS), bn1_ss);

  attn_softmax_ctx<<<dim3(SS / 64, HH, BB), 128, 0, stream>>>(qb, kTb, vb, bn1_ss, ctx);

  ctx_stats<<<(M * DD) / 1024, 256, 0, stream>>>(ctx, bn2_sum, bn2_sq);
  bn_finalize<<<1, 32, 0, stream>>>(bn2_sum, bn2_sq, bn2_g, bn2_b,
                                    1.0f / ((float)BB * SS * HD), bn2_ss);
  ctx_norm_transpose<<<(M * DD) / 256, 256, 0, stream>>>(ctx, bn2_ss, attnb);

  gemm_wmma<1><<<dim3(M / 64, DD / 64), 128, 0, stream>>>(attnb, wob, bo, x, x2, nullptr, M, DD, DD);
  ln_cast<<<M, 256, 0, stream>>>(x2, ln2_g, ln2_b, y2b);
  gemm_wmma<2><<<dim3(M / 64, (4 * DD) / 64), 128, 0, stream>>>(y2b, w1b, b1, nullptr, nullptr, h1b, M, 4 * DD, DD);
  gemm_wmma<3><<<dim3(M / 64, DD / 64), 128, 0, stream>>>(h1b, w2b, b2, x2, out, nullptr, M, DD, 4 * DD);

  (void)in_sizes; (void)n_in; (void)out_size; (void)ws_size;
}
